// VimMamba_36971078483993
// MI455X (gfx1250) — matrix operand order repaired and checked by reading, not compiled, not run on hardware
//
#include <hip/hip_runtime.h>

// ---- problem constants -----------------------------------------------------
#define B_SZ   8
#define LSEQ   2048
#define DMODEL 384
#define DSTATE 16
#define DCONV  4
#define DINNER 768
#define DTRANK 24
#define NROWS  (B_SZ * LSEQ)          // 16384 token rows

// ---- WMMA types ------------------------------------------------------------
typedef __attribute__((ext_vector_type(16))) __bf16 v16bf;
typedef __attribute__((ext_vector_type(8)))  float  v8f;

union FragBF {
    v16bf v;
    unsigned int ui[8];
};

__device__ __forceinline__ unsigned short f2bf(float f) {
    unsigned int x = __float_as_uint(f);
    unsigned int r = x + 0x7fffu + ((x >> 16) & 1u);   // round-to-nearest-even
    return (unsigned short)(r >> 16);
}

// ---- fragment loaders (contiguous k = 16*(lane/16) + s for A and B, wave32) -------------------
// A: 16x32 bf16, row-major source A[row, k], lda in elements.
__device__ __forceinline__ void load_a(FragBF& a, const unsigned short* __restrict__ A,
                                       int rowBase, int lda, int kbase, int lane) {
    int m = lane & 15, g = lane >> 4;
    const unsigned short* base = A + (size_t)(rowBase + m) * lda + kbase;
#pragma unroll
    for (int v = 0; v < 8; ++v) {
        int k0 = g * 16 + v * 2;   // {0..30} even
        a.ui[v] = *(const unsigned int*)(base + k0);
    }
}

// A with K guard (pairs are even-aligned; pair valid iff k0+1 < kmax)
__device__ __forceinline__ void load_a_kg(FragBF& a, const unsigned short* __restrict__ A,
                                          int rowBase, int lda, int kbase, int kmax, int lane) {
    int m = lane & 15, g = lane >> 4;
    const unsigned short* base = A + (size_t)(rowBase + m) * lda;
#pragma unroll
    for (int v = 0; v < 8; ++v) {
        int k0 = kbase + g * 16 + v * 2;
        a.ui[v] = (k0 + 1 < kmax) ? *(const unsigned int*)(base + k0) : 0u;
    }
}

// B: 32x16 bf16 where B[k,n] = W[n,k], W row-major [N,K], ldw in elements.
__device__ __forceinline__ void load_b(FragBF& b, const unsigned short* __restrict__ W,
                                       int colBase, int ldw, int kbase, int lane) {
    int n = lane & 15, g = lane >> 4;
    const unsigned short* base = W + (size_t)(colBase + n) * ldw + kbase;
#pragma unroll
    for (int v = 0; v < 8; ++v) {
        int k0 = g * 16 + v * 2;
        b.ui[v] = *(const unsigned int*)(base + k0);
    }
}

// B with N guard (rows of W beyond nmax -> zeros)
__device__ __forceinline__ void load_b_ng(FragBF& b, const unsigned short* __restrict__ W,
                                          int colBase, int ldw, int kbase, int nmax, int lane) {
    int n = lane & 15, g = lane >> 4;
    bool ok = (colBase + n) < nmax;
    const unsigned short* base = W + (size_t)(colBase + n) * ldw + kbase;
#pragma unroll
    for (int v = 0; v < 8; ++v) {
        int k0 = g * 16 + v * 2;
        b.ui[v] = ok ? *(const unsigned int*)(base + k0) : 0u;
    }
}

// B with K guard
__device__ __forceinline__ void load_b_kg(FragBF& b, const unsigned short* __restrict__ W,
                                          int colBase, int ldw, int kbase, int kmax, int lane) {
    int n = lane & 15, g = lane >> 4;
    const unsigned short* base = W + (size_t)(colBase + n) * ldw;
#pragma unroll
    for (int v = 0; v < 8; ++v) {
        int k0 = kbase + g * 16 + v * 2;
        b.ui[v] = (k0 + 1 < kmax) ? *(const unsigned int*)(base + k0) : 0u;
    }
}

#define WMMA_BF16(c, a, b) \
    __builtin_amdgcn_wmma_f32_16x16x32_bf16(false, (a).v, false, (b).v, (short)0, (c), false, false)

// ---- kernel 0: f32 -> bf16 bits --------------------------------------------
__global__ void k_f2bf(const float* __restrict__ src, unsigned short* __restrict__ dst, int n) {
    int i = blockIdx.x * 256 + threadIdx.x;
    if (i < n) dst[i] = f2bf(src[i]);
}

// ---- kernel 1: in_proj GEMM [16384,384]x[1536,384]^T -> x,z in [B,d,L] -----
__global__ void k_inproj(const unsigned short* __restrict__ A,   // hidden bf16 [NROWS, DMODEL]
                         const unsigned short* __restrict__ W,   // in_proj_w bf16 [1536, DMODEL]
                         float* __restrict__ xbuf,               // [B, DINNER, L]
                         float* __restrict__ zbuf) {             // [B, DINNER, L]
    const int lane = threadIdx.x & 31;
    const int wid  = threadIdx.x >> 5;
    const int NT = (2 * DINNER) / 16;                            // 96
    int tile = blockIdx.x * 8 + wid;
    int tm = tile / NT, tn = tile % NT;
    int rowBase = tm * 16, colBase = tn * 16;

    v8f c = {0.f, 0.f, 0.f, 0.f, 0.f, 0.f, 0.f, 0.f};
    FragBF a, b;
    for (int kb = 0; kb < DMODEL; kb += 32) {
        load_a(a, A, rowBase, DMODEL, kb, lane);
        load_b(b, W, colBase, DMODEL, kb, lane);
        c = WMMA_BF16(c, a, b);
    }
    int n = lane & 15, g = lane >> 4;
    int col = colBase + n;
#pragma unroll
    for (int r = 0; r < 8; ++r) {
        int row = rowBase + r + g * 8;
        int bi = row >> 11;            // / LSEQ
        int t  = row & (LSEQ - 1);
        float v = c[r];
        if (col < DINNER) xbuf[((size_t)bi * DINNER + col) * LSEQ + t] = v;
        else              zbuf[((size_t)bi * DINNER + (col - DINNER)) * LSEQ + t] = v;
    }
}

// ---- kernel 2: fused causal depthwise conv (fwd + flipped bwd) -------------
__global__ void k_conv(const float* __restrict__ x,              // [B, DINNER, L]
                       const float* __restrict__ cw,  const float* __restrict__ cb,
                       const float* __restrict__ cwb, const float* __restrict__ cbb,
                       float* __restrict__ xc_f, float* __restrict__ xc_b,          // [B,d,L]
                       unsigned short* __restrict__ xcT_f,
                       unsigned short* __restrict__ xcT_b) {                        // [B*L, d]
    size_t idx = (size_t)blockIdx.x * 256 + threadIdx.x;
    if (idx >= (size_t)B_SZ * DINNER * LSEQ) return;
    int t = idx & (LSEQ - 1);
    int d = (int)((idx >> 11) % DINNER);
    int bi = (int)(idx / ((size_t)DINNER * LSEQ));
    const float* xr = x + ((size_t)bi * DINNER + d) * LSEQ;

    float acc = cb[d];
#pragma unroll
    for (int k = 0; k < DCONV; ++k) {
        int tt = t - (DCONV - 1) + k;
        if (tt >= 0) acc += cw[d * DCONV + k] * xr[tt];
    }
    xc_f[idx] = acc;
    xcT_f[((size_t)bi * LSEQ + t) * DINNER + d] = f2bf(acc);

    float accb = cbb[d];                                         // operates on flipped x
#pragma unroll
    for (int k = 0; k < DCONV; ++k) {
        int tt = t - (DCONV - 1) + k;
        if (tt >= 0) accb += cwb[d * DCONV + k] * xr[(LSEQ - 1) - tt];
    }
    xc_b[idx] = accb;
    xcT_b[((size_t)bi * LSEQ + t) * DINNER + d] = f2bf(accb);
}

// ---- kernel 3: x_proj GEMM [16384,768]x[56,768]^T (N padded to 64) ---------
__global__ void k_xproj(const unsigned short* __restrict__ A,    // xcT bf16 [NROWS, DINNER]
                        const unsigned short* __restrict__ W,    // x_proj_w bf16 [56, DINNER]
                        unsigned short* __restrict__ dtlow,      // bf16 [NROWS, DTRANK]
                        float* __restrict__ Bm,                  // [NROWS, DSTATE]
                        float* __restrict__ Cm) {                // [NROWS, DSTATE]
    const int lane = threadIdx.x & 31;
    const int wid  = threadIdx.x >> 5;
    const int NT = 64 / 16;                                      // 4 (padded)
    const int NVAL = DTRANK + 2 * DSTATE;                        // 56
    int tile = blockIdx.x * 8 + wid;
    int tm = tile / NT, tn = tile % NT;
    int rowBase = tm * 16, colBase = tn * 16;

    v8f c = {0.f, 0.f, 0.f, 0.f, 0.f, 0.f, 0.f, 0.f};
    FragBF a, b;
    for (int kb = 0; kb < DINNER; kb += 32) {
        load_a(a, A, rowBase, DINNER, kb, lane);
        load_b_ng(b, W, colBase, DINNER, kb, NVAL, lane);
        c = WMMA_BF16(c, a, b);
    }
    int n = lane & 15, g = lane >> 4;
    int col = colBase + n;
#pragma unroll
    for (int r = 0; r < 8; ++r) {
        int row = rowBase + r + g * 8;
        float v = c[r];
        if (col < DTRANK)                 dtlow[(size_t)row * DTRANK + col] = f2bf(v);
        else if (col < DTRANK + DSTATE)   Bm[(size_t)row * DSTATE + (col - DTRANK)] = v;
        else if (col < NVAL)              Cm[(size_t)row * DSTATE + (col - DTRANK - DSTATE)] = v;
    }
}

// ---- kernel 4: dt_proj GEMM [16384,24]x[768,24]^T (K padded to 32) ---------
__global__ void k_dtproj(const unsigned short* __restrict__ A,   // dtlow bf16 [NROWS, DTRANK]
                         const unsigned short* __restrict__ W,   // dt_proj_w bf16 [DINNER, DTRANK]
                         float* __restrict__ dt_out) {           // [NROWS, DINNER]
    const int lane = threadIdx.x & 31;
    const int wid  = threadIdx.x >> 5;
    const int NT = DINNER / 16;                                  // 48
    int tile = blockIdx.x * 8 + wid;
    int tm = tile / NT, tn = tile % NT;
    int rowBase = tm * 16, colBase = tn * 16;

    v8f c = {0.f, 0.f, 0.f, 0.f, 0.f, 0.f, 0.f, 0.f};
    FragBF a, b;
    load_a_kg(a, A, rowBase, DTRANK, 0, DTRANK, lane);
    load_b_kg(b, W, colBase, DTRANK, 0, DTRANK, lane);
    c = WMMA_BF16(c, a, b);

    int n = lane & 15, g = lane >> 4;
    int col = colBase + n;
#pragma unroll
    for (int r = 0; r < 8; ++r) {
        int row = rowBase + r + g * 8;
        dt_out[(size_t)row * DINNER + col] = c[r];
    }
}

// ---- kernel 5: selective scan, both directions in one grid -----------------
// block = 256 threads = one (branch, batch, 256-channel chunk); 48 blocks.
// B(t)/C(t) rows are DMA'd into LDS with GLOBAL_LOAD_ASYNC_TO_LDS_B128
// (ASYNCcnt-tracked), double buffered: step t+1 streams in while t computes.
__global__ void k_scan(const float* __restrict__ u0,  const float* __restrict__ dt0,
                       const float* __restrict__ B0,  const float* __restrict__ C0,
                       const float* __restrict__ Al0, const float* __restrict__ D0,
                       const float* __restrict__ bia0, float* __restrict__ y0,
                       const float* __restrict__ u1,  const float* __restrict__ dt1,
                       const float* __restrict__ B1,  const float* __restrict__ C1,
                       const float* __restrict__ Al1, const float* __restrict__ D1,
                       const float* __restrict__ bia1, float* __restrict__ y1) {
    __shared__ float sBC[2][2 * DSTATE];      // [buf][ B:0..15 | C:16..31 ]
    int tid = threadIdx.x;
    int blk = blockIdx.x;
    int branch = (blk >= 24) ? 1 : 0;
    blk -= branch * 24;
    int bi = blk / 3, chunk = blk % 3;
    int d = chunk * 256 + tid;

    const float* u_   = branch ? u1  : u0;
    const float* dt_  = branch ? dt1 : dt0;
    const float* Bm   = branch ? B1  : B0;
    const float* Cm   = branch ? C1  : C0;
    const float* Alog = branch ? Al1 : Al0;
    const float* Dv   = branch ? D1  : D0;
    const float* bia  = branch ? bia1: bia0;
    float*       y    = branch ? y1  : y0;

    // LDS byte address of sBC (generic LDS pointers carry DS offset in low 32b)
    unsigned ldsBase = (unsigned)(size_t)(&sBC[0][0]);

    // 8 lanes of wave 0 each async-copy 16B: lanes 0-3 -> B row, 4-7 -> C row.
    auto issue = [&](int t, int buf) {
        if (tid < 8) {
            int part = tid >> 2;              // 0 = B, 1 = C
            int q = tid & 3;
            const float* src = (part ? Cm : Bm) +
                               ((size_t)bi * LSEQ + t) * DSTATE + q * 4;
            unsigned dst = ldsBase + (unsigned)((buf * 2 * DSTATE + part * DSTATE + q * 4) * 4);
            asm volatile("global_load_async_to_lds_b128 %0, %1, off"
                         :: "v"(dst), "v"((unsigned long long)(size_t)src)
                         : "memory");
        }
    };

    float Arow[DSTATE], h[DSTATE];
#pragma unroll
    for (int n = 0; n < DSTATE; ++n) {
        Arow[n] = -expf(Alog[d * DSTATE + n]);
        h[n] = 0.f;
    }
    float bias = bia[d], Dd = Dv[d];
    const float* ur = u_ + ((size_t)bi * DINNER + d) * LSEQ;
    float* yr = y + ((size_t)bi * DINNER + d) * LSEQ;

    issue(0, 0);                              // prime buffer 0

    for (int t = 0; t < LSEQ; ++t) {
        int buf = t & 1;
        asm volatile("s_wait_asynccnt 0x0" ::: "memory");  // buf[t&1] resident
        __syncthreads();
        if (t + 1 < LSEQ) issue(t + 1, buf ^ 1);           // prefetch next step

        const float* sB = sBC[buf];
        const float* sC = sBC[buf] + DSTATE;

        float draw  = dt_[((size_t)bi * LSEQ + t) * DINNER + d] + bias;
        float delta = (draw > 20.f) ? draw : log1pf(expf(draw));   // softplus
        float uu = ur[t];
        float du = delta * uu;
        float yt = 0.f;
#pragma unroll
        for (int n = 0; n < DSTATE; ++n) {
            h[n] = expf(delta * Arow[n]) * h[n] + du * sB[n];
            yt += h[n] * sC[n];
        }
        yr[t] = yt + uu * Dd;
        __syncthreads();                      // reads done before buf reuse
    }
}

// ---- kernel 6: combine + flip + SiLU gate -> bf16 [B*L, d] -----------------
__global__ void k_gate(const float* __restrict__ yf, const float* __restrict__ yb,
                       const float* __restrict__ z, unsigned short* __restrict__ yg) {
    size_t idx = (size_t)blockIdx.x * 256 + threadIdx.x;
    if (idx >= (size_t)B_SZ * DINNER * LSEQ) return;
    int t = idx & (LSEQ - 1);
    int d = (int)((idx >> 11) % DINNER);
    int bi = (int)(idx / ((size_t)DINNER * LSEQ));
    size_t base = ((size_t)bi * DINNER + d) * LSEQ;
    float v  = yf[base + t] + yb[base + (LSEQ - 1 - t)];
    float zz = z[base + t];
    float g  = zz / (1.f + expf(-zz));                             // silu
    yg[((size_t)bi * LSEQ + t) * DINNER + d] = f2bf(v * g);
}

// ---- kernel 7: out_proj GEMM [16384,768]x[384,768]^T -> d_out --------------
__global__ void k_outproj(const unsigned short* __restrict__ A,  // yg bf16 [NROWS, DINNER]
                          const unsigned short* __restrict__ W,  // out_proj_w bf16 [DMODEL, DINNER]
                          float* __restrict__ out) {             // [NROWS, DMODEL]
    const int lane = threadIdx.x & 31;
    const int wid  = threadIdx.x >> 5;
    const int NT = DMODEL / 16;                                  // 24
    int tile = blockIdx.x * 8 + wid;
    int tm = tile / NT, tn = tile % NT;
    int rowBase = tm * 16, colBase = tn * 16;

    v8f c = {0.f, 0.f, 0.f, 0.f, 0.f, 0.f, 0.f, 0.f};
    FragBF a, b;
    for (int kb = 0; kb < DINNER; kb += 32) {
        load_a(a, A, rowBase, DINNER, kb, lane);
        load_b(b, W, colBase, DINNER, kb, lane);
        c = WMMA_BF16(c, a, b);
    }
    int n = lane & 15, g = lane >> 4;
    int col = colBase + n;
#pragma unroll
    for (int r = 0; r < 8; ++r) {
        int row = rowBase + r + g * 8;
        out[(size_t)row * DMODEL + col] = c[r];
    }
}

// ---- host ------------------------------------------------------------------
extern "C" void kernel_launch(void* const* d_in, const int* in_sizes, int n_in,
                              void* d_out, int out_size, void* d_ws, size_t ws_size,
                              hipStream_t stream) {
    (void)in_sizes; (void)n_in; (void)out_size; (void)ws_size;

    const float* hidden   = (const float*)d_in[0];
    const float* in_w     = (const float*)d_in[1];
    const float* conv_w   = (const float*)d_in[2];
    const float* conv_b   = (const float*)d_in[3];
    const float* xp_w     = (const float*)d_in[4];
    const float* dtp_w    = (const float*)d_in[5];
    const float* dtp_b    = (const float*)d_in[6];
    const float* A_log    = (const float*)d_in[7];
    const float* Dv       = (const float*)d_in[8];
    const float* conv_w_b = (const float*)d_in[9];
    const float* conv_b_b = (const float*)d_in[10];
    const float* xp_w_b   = (const float*)d_in[11];
    const float* dtp_w_b  = (const float*)d_in[12];
    const float* dtp_b_b  = (const float*)d_in[13];
    const float* A_log_b  = (const float*)d_in[14];
    const float* Dv_b     = (const float*)d_in[15];
    const float* out_w    = (const float*)d_in[16];
    float* out = (float*)d_out;

    char* ws = (char*)d_ws;
    size_t off = 0;
    auto alloc = [&](size_t bytes) -> void* {
        void* p = ws + off;
        off = (off + bytes + 255) & ~(size_t)255;
        return p;
    };

    unsigned short* hid_bf  = (unsigned short*)alloc((size_t)NROWS * DMODEL * 2);
    unsigned short* winp_bf = (unsigned short*)alloc((size_t)2 * DINNER * DMODEL * 2);
    unsigned short* wxpf_bf = (unsigned short*)alloc((size_t)56 * DINNER * 2);
    unsigned short* wxpb_bf = (unsigned short*)alloc((size_t)56 * DINNER * 2);
    unsigned short* wdtf_bf = (unsigned short*)alloc((size_t)DINNER * DTRANK * 2);
    unsigned short* wdtb_bf = (unsigned short*)alloc((size_t)DINNER * DTRANK * 2);
    unsigned short* wout_bf = (unsigned short*)alloc((size_t)DMODEL * DINNER * 2);

    float* xbuf  = (float*)alloc((size_t)B_SZ * DINNER * LSEQ * 4);
    float* zbuf  = (float*)alloc((size_t)B_SZ * DINNER * LSEQ * 4);
    float* xc_f  = (float*)alloc((size_t)B_SZ * DINNER * LSEQ * 4);
    float* xc_b  = (float*)alloc((size_t)B_SZ * DINNER * LSEQ * 4);
    unsigned short* xcT_f = (unsigned short*)alloc((size_t)NROWS * DINNER * 2);
    unsigned short* xcT_b = (unsigned short*)alloc((size_t)NROWS * DINNER * 2);
    unsigned short* dtlow_f = (unsigned short*)alloc((size_t)NROWS * DTRANK * 2);
    unsigned short* dtlow_b = (unsigned short*)alloc((size_t)NROWS * DTRANK * 2);
    float* Bm_f = (float*)alloc((size_t)NROWS * DSTATE * 4);
    float* Cm_f = (float*)alloc((size_t)NROWS * DSTATE * 4);
    float* Bm_b = (float*)alloc((size_t)NROWS * DSTATE * 4);
    float* Cm_b = (float*)alloc((size_t)NROWS * DSTATE * 4);
    float* dt_f = (float*)alloc((size_t)NROWS * DINNER * 4);
    float* dt_b = (float*)alloc((size_t)NROWS * DINNER * 4);
    float* y_f  = (float*)alloc((size_t)B_SZ * DINNER * LSEQ * 4);
    float* y_b  = (float*)alloc((size_t)B_SZ * DINNER * LSEQ * 4);
    unsigned short* yg = (unsigned short*)alloc((size_t)NROWS * DINNER * 2);

    auto cvt = [&](const float* s, unsigned short* d, int n) {
        k_f2bf<<<(n + 255) / 256, 256, 0, stream>>>(s, d, n);
    };
    cvt(hidden,  hid_bf,  NROWS * DMODEL);
    cvt(in_w,    winp_bf, 2 * DINNER * DMODEL);
    cvt(xp_w,    wxpf_bf, 56 * DINNER);
    cvt(xp_w_b,  wxpb_bf, 56 * DINNER);
    cvt(dtp_w,   wdtf_bf, DINNER * DTRANK);
    cvt(dtp_w_b, wdtb_bf, DINNER * DTRANK);
    cvt(out_w,   wout_bf, DMODEL * DINNER);

    // in_proj: (16384/16)*(1536/16) = 98304 tiles, 8 waves/block
    k_inproj<<<(NROWS / 16) * (2 * DINNER / 16) / 8, 256, 0, stream>>>(hid_bf, winp_bf, xbuf, zbuf);

    // conv (both directions)
    {
        size_t n = (size_t)B_SZ * DINNER * LSEQ;
        k_conv<<<(unsigned)((n + 255) / 256), 256, 0, stream>>>(
            xbuf, conv_w, conv_b, conv_w_b, conv_b_b, xc_f, xc_b, xcT_f, xcT_b);
    }

    // x_proj: (16384/16)*(64/16) = 4096 tiles
    k_xproj<<<(NROWS / 16) * 4 / 8, 256, 0, stream>>>(xcT_f, wxpf_bf, dtlow_f, Bm_f, Cm_f);
    k_xproj<<<(NROWS / 16) * 4 / 8, 256, 0, stream>>>(xcT_b, wxpb_bf, dtlow_b, Bm_b, Cm_b);

    // dt_proj: (16384/16)*(768/16) = 49152 tiles
    k_dtproj<<<(NROWS / 16) * (DINNER / 16) / 8, 256, 0, stream>>>(dtlow_f, wdtf_bf, dt_f);
    k_dtproj<<<(NROWS / 16) * (DINNER / 16) / 8, 256, 0, stream>>>(dtlow_b, wdtb_bf, dt_b);

    // selective scan, both branches in one grid (48 blocks x 256)
    k_scan<<<48, 256, 0, stream>>>(xc_f, dt_f, Bm_f, Cm_f, A_log, Dv, dtp_b, y_f,
                                   xc_b, dt_b, Bm_b, Cm_b, A_log_b, Dv_b, dtp_b_b, y_b);

    // gate
    {
        size_t n = (size_t)B_SZ * DINNER * LSEQ;
        k_gate<<<(unsigned)((n + 255) / 256), 256, 0, stream>>>(y_f, y_b, zbuf, yg);
    }

    // out_proj: (16384/16)*(384/16) = 24576 tiles
    k_outproj<<<(NROWS / 16) * (DMODEL / 16) / 8, 256, 0, stream>>>(yg, wout_bf, out);
}
